// RCA_85985245266561
// MI455X (gfx1250) — matrix operand order repaired and checked by reading, not compiled, not run on hardware
//
#include <hip/hip_runtime.h>

#define BN   8
#define TN   20
#define VN   6
#define CN   512
#define HWN  49
#define NCLS 8
#define FLOC (CN * HWN)    // 25088
#define FGLB (2 * FLOC)    // 50176
#define ROWS (BN * TN)     // 160
#define NSPLIT 4
#define PSTRIDE ((size_t)ROWS * VN * CN)   // one split-K partial plane

typedef __attribute__((ext_vector_type(16))) __bf16 bf16x16;
typedef __attribute__((ext_vector_type(8)))  float  f32x8;
typedef __attribute__((ext_vector_type(4)))  float  f32x4;

__device__ __forceinline__ float sigmoidf_(float x) {
    return 1.0f / (1.0f + __expf(-x));
}

// Load one 16x32 bf16 A-fragment (lane L: row m = L%16, K runs kb2..kb2+7 and
// 16+kb2..16+kb2+7, kb2 = L<16 ? 0 : 8) -> 4x global_load_b128 + packed cvt.
__device__ __forceinline__ bf16x16 load_a_frag(const float* base, int kb2) {
    const f32x4* p = (const f32x4*)(base + kb2);
    f32x4 x0 = p[0], x1 = p[1];
    const f32x4* q = (const f32x4*)(base + 16 + kb2);
    f32x4 y0 = q[0], y1 = q[1];
    bf16x16 a;
    a[0]  = (__bf16)x0.x; a[1]  = (__bf16)x0.y; a[2]  = (__bf16)x0.z; a[3]  = (__bf16)x0.w;
    a[4]  = (__bf16)x1.x; a[5]  = (__bf16)x1.y; a[6]  = (__bf16)x1.z; a[7]  = (__bf16)x1.w;
    a[8]  = (__bf16)y0.x; a[9]  = (__bf16)y0.y; a[10] = (__bf16)y0.z; a[11] = (__bf16)y0.w;
    a[12] = (__bf16)y1.x; a[13] = (__bf16)y1.y; a[14] = (__bf16)y1.z; a[15] = (__bf16)y1.w;
    return a;
}

// Stage a 64(K) x 16(N) f32 weight tile into LDS as bf16, transposed [n][k].
// Threads 0..127: two float4 loads each (non-temporal: weights are single-use,
// keep L2 for feats/agg). Fixed trip count -> clean global_load_b128 codegen.
__device__ __forceinline__ void stage_tile(__bf16 (*bs)[66],
                                           const float* __restrict__ wbase, // Wv + kk0*CN + n0
                                           int tid) {
    if (tid < 128) {
        #pragma unroll
        for (int j = 0; j < 2; ++j) {
            int idx = tid + 128 * j;          // 0..255 quads
            int kk  = idx >> 2;               // 0..63
            int nn4 = (idx & 3) << 2;         // 0,4,8,12
            f32x4 wv4 = __builtin_nontemporal_load(
                (const f32x4*)(wbase + (size_t)kk * CN + nn4));
            bs[nn4 + 0][kk] = (__bf16)wv4.x;
            bs[nn4 + 1][kk] = (__bf16)wv4.y;
            bs[nn4 + 2][kk] = (__bf16)wv4.z;
            bs[nn4 + 3][kk] = (__bf16)wv4.w;
        }
    }
}

// Split-K GEMM partials: part[ks][row][v][c] += A[row, kbeg:kbeg+Kc] @ W[v][.., c]
// A element k < FLOC -> feats, else agg (global pass). grid = (CN/16, VN, NSPLIT),
// block = 160 = 5 waves; wave w owns row tiles {2w, 2w+1}; double-buffered LDS
// B strip, one barrier per 64-wide K-step, 4 WMMAs per step.
__global__ __launch_bounds__(160)
void gemm_part_kernel(const float* __restrict__ feats,
                      const float* __restrict__ aggp,   // may be null (local pass)
                      const float* __restrict__ Wmat,   // [VN][K][CN]
                      float* __restrict__ part,         // [NSPLIT][ROWS][VN][CN]
                      int K, int Kc)
{
    const int n0   = blockIdx.x * 16;
    const int v    = blockIdx.y;
    const int ks   = blockIdx.z;
    const int kbeg = ks * Kc;
    const int tid  = threadIdx.x;
    const int wv   = tid >> 5;
    const int lane = tid & 31;

    __shared__ __bf16 Bs[2][16][66];

    const float* Wv = Wmat + (size_t)v * K * CN + n0;

    const int m   = lane & 15;
    const int kb2 = (lane < 16) ? 0 : 8;     // A K-offset
    const int kbB = (lane < 16) ? 0 : 16;    // B K-offset
    const int n   = lane & 15;

    const int row0 = wv * 32 + m;
    const int row1 = row0 + 16;

    const float* fr0 = feats + ((size_t)row0 * VN + v) * FLOC;
    const float* fr1 = feats + ((size_t)row1 * VN + v) * FLOC;
    const float* ar0 = aggp ? (aggp + (size_t)row0 * FLOC) : fr0;
    const float* ar1 = aggp ? (aggp + (size_t)row1 * FLOC) : fr1;

    f32x8 acc0 = {};
    f32x8 acc1 = {};

    const int nIter = Kc >> 6;               // Kc / 64, exact
    stage_tile(Bs[0], Wv + (size_t)kbeg * CN, tid);
    __syncthreads();

    for (int it = 0; it < nIter; ++it) {
        const int k0 = kbeg + (it << 6);
        if (it + 1 < nIter)
            stage_tile(Bs[(it + 1) & 1], Wv + (size_t)(k0 + 64) * CN, tid);
        const int cur = it & 1;

        #pragma unroll
        for (int s = 0; s < 2; ++s) {
            const int kk = k0 + s * 32;
            bf16x16 bfrag;
            #pragma unroll
            for (int e = 0; e < 16; ++e) bfrag[e] = Bs[cur][n][s * 32 + kb2 + e + (e & 8)];

            const float* a0 = (kk < FLOC) ? (fr0 + kk) : (ar0 + (kk - FLOC));
            const float* a1 = (kk < FLOC) ? (fr1 + kk) : (ar1 + (kk - FLOC));
            bf16x16 af0 = load_a_frag(a0, kb2);
            bf16x16 af1 = load_a_frag(a1, kb2);

            acc0 = __builtin_amdgcn_wmma_f32_16x16x32_bf16(
                false, af0, false, bfrag, (short)0, acc0, false, false);
            acc1 = __builtin_amdgcn_wmma_f32_16x16x32_bf16(
                false, af1, false, bfrag, (short)0, acc1, false, false);
        }
        __syncthreads();
    }

    // C/D layout: VGPR rr -> M = rr + (lane<16?0:8), N = lane%16
    float* pout = part + (size_t)ks * PSTRIDE;
    const int moff = (lane < 16) ? 0 : 8;
    #pragma unroll
    for (int rr = 0; rr < 8; ++rr) {
        int g0 = wv * 32 + rr + moff;
        int g1 = g0 + 16;
        pout[((size_t)g0 * VN + v) * CN + n0 + n] = acc0[rr];
        pout[((size_t)g1 * VN + v) * CN + n0 + n] = acc1[rr];
    }
}

// outw[i] = sigmoid(sum_ks part[ks][i] + bias[v][c])  (deterministic combine)
__global__ __launch_bounds__(256)
void combine_sig_kernel(const float* __restrict__ part,
                        const float* __restrict__ bias,
                        float* __restrict__ outw)
{
    int idx = blockIdx.x * blockDim.x + threadIdx.x;
    if (idx >= (int)PSTRIDE) return;
    int c = idx % CN;
    int v = (idx / CN) % VN;
    float s = part[idx] + part[idx + PSTRIDE]
            + part[idx + 2 * PSTRIDE] + part[idx + 3 * PSTRIDE];
    outw[idx] = sigmoidf_(s + bias[v * CN + c]);
}

// One wave per (bt,v) row: optionally fw = w*r (stored), then sum over CN.
__global__ __launch_bounds__(256)
void rowsum_kernel(const float* __restrict__ wbuf,
                   const float* __restrict__ rbuf,   // may be null
                   float* __restrict__ fwbuf,        // used iff rbuf != null
                   float* __restrict__ sums)         // [ROWS*VN]
{
    int g    = blockIdx.x * (blockDim.x >> 5) + (threadIdx.x >> 5);
    int lane = threadIdx.x & 31;
    if (g >= ROWS * VN) return;
    const size_t base = (size_t)g * CN;
    float s = 0.0f;
    for (int c = lane; c < CN; c += 32) {
        float x = wbuf[base + c];
        if (rbuf) { x *= rbuf[base + c]; fwbuf[base + c] = x; }
        s += x;
    }
    #pragma unroll
    for (int off = 16; off > 0; off >>= 1) s += __shfl_xor(s, off, 32);
    if (lane == 0) sums[g] = s;
}

// out[r,c,hw] = sum_v feats[r,v,c,hw] * wbuf[r,v,c] / sums[r,v]
__global__ __launch_bounds__(256)
void weighted_agg_kernel(const float* __restrict__ feats,
                         const float* __restrict__ wbuf,
                         const float* __restrict__ sums,
                         float* __restrict__ outagg)   // [ROWS][FLOC]
{
    int idx = blockIdx.x * blockDim.x + threadIdx.x;
    const int total = ROWS * CN * HWN;
    if (idx >= total) return;
    int hw = idx % HWN;
    int c  = (idx / HWN) % CN;
    int r  = idx / (HWN * CN);
    float acc = 0.0f;
    #pragma unroll
    for (int v = 0; v < VN; ++v) {
        float wn = wbuf[((size_t)r * VN + v) * CN + c] / sums[r * VN + v];
        acc += feats[(((size_t)r * VN + v) * CN + c) * HWN + hw] * wn;
    }
    outagg[idx] = acc;
}

// frame_out[r, :8] = cat(refined[r], agg[r]) @ W_final + b_final
__global__ __launch_bounds__(256)
void final_proj_kernel(const float* __restrict__ refined,
                       const float* __restrict__ aggp,
                       const float* __restrict__ Wf,   // [FGLB][8]
                       const float* __restrict__ bf,   // [8]
                       float* __restrict__ frame_out)  // [ROWS][8]
{
    int r   = blockIdx.x;
    int tid = threadIdx.x;
    float acc[8] = {0, 0, 0, 0, 0, 0, 0, 0};
    for (int f = tid; f < FGLB; f += 256) {
        float x = (f < FLOC) ? refined[(size_t)r * FLOC + f]
                             : aggp[(size_t)r * FLOC + (f - FLOC)];
        const float* wrow = Wf + (size_t)f * 8;
        #pragma unroll
        for (int j = 0; j < 8; ++j) acc[j] += x * wrow[j];
    }
    __shared__ float red[256];
    for (int j = 0; j < 8; ++j) {
        red[tid] = acc[j];
        __syncthreads();
        for (int s = 128; s > 0; s >>= 1) {
            if (tid < s) red[tid] += red[tid + s];
            __syncthreads();
        }
        if (tid == 0) frame_out[r * 8 + j] = red[0] + bf[j];
        __syncthreads();
    }
}

// out[b, :8] = frame_out[b].reshape(T*NC) @ W_time + b_time   (64 outputs)
__global__ void time_head_kernel(const float* __restrict__ frame_out,
                                 const float* __restrict__ Wt,   // [T*NC][NC]
                                 const float* __restrict__ bt,   // [NC]
                                 float* __restrict__ out)        // [BN][NC]
{
    int tid = threadIdx.x;
    if (tid >= BN * NCLS) return;
    int b = tid >> 3, j = tid & 7;
    float acc = bt[j];
    for (int k = 0; k < TN * NCLS; ++k)
        acc += frame_out[b * TN * NCLS + k] * Wt[k * NCLS + j];
    out[b * NCLS + j] = acc;
}

extern "C" void kernel_launch(void* const* d_in, const int* in_sizes, int n_in,
                              void* d_out, int out_size, void* d_ws, size_t ws_size,
                              hipStream_t stream)
{
    const float* feats    = (const float*)d_in[0];
    const float* W_local  = (const float*)d_in[1];
    const float* b_local  = (const float*)d_in[2];
    const float* W_global = (const float*)d_in[3];
    const float* b_global = (const float*)d_in[4];
    const float* W_final  = (const float*)d_in[5];
    const float* b_final  = (const float*)d_in[6];
    const float* W_time   = (const float*)d_in[7];
    const float* b_time   = (const float*)d_in[8];
    float* out = (float*)d_out;

    float* ws      = (float*)d_ws;
    float* w_sig   = ws;                                   // 960*512
    float* r_sig   = w_sig   + PSTRIDE;                    // 960*512
    float* fw      = r_sig   + PSTRIDE;                    // 960*512
    float* s1      = fw      + PSTRIDE;                    // 960
    float* s2      = s1      + ROWS * VN;                  // 960
    float* aggb    = s2      + ROWS * VN;                  // 160*25088
    float* refined = aggb    + (size_t)ROWS * FLOC;        // 160*25088
    float* frame   = refined + (size_t)ROWS * FLOC;        // 160*8
    float* part    = frame   + ROWS * NCLS;                // 4*960*512

    dim3 gemmGrid(CN / 16, VN, NSPLIT);
    const int totalAgg = ROWS * CN * HWN;
    const int combBlocks = ((int)PSTRIDE + 255) / 256;

    // 1) local attention logits (split-K partials) -> combine + sigmoid
    gemm_part_kernel<<<gemmGrid, 160, 0, stream>>>(feats, nullptr, W_local, part,
                                                   FLOC, FLOC / NSPLIT);
    combine_sig_kernel<<<combBlocks, 256, 0, stream>>>(part, b_local, w_sig);
    // 2) channel sums of w
    rowsum_kernel<<<(ROWS * VN + 7) / 8, 256, 0, stream>>>(w_sig, nullptr, nullptr, s1);
    // 3) agg = sum_v feats * w_norm
    weighted_agg_kernel<<<(totalAgg + 255) / 256, 256, 0, stream>>>(feats, w_sig, s1, aggb);
    // 4) global refinement logits (A = [feats | agg]) -> combine + sigmoid
    gemm_part_kernel<<<gemmGrid, 160, 0, stream>>>(feats, aggb, W_global, part,
                                                   FGLB, FGLB / NSPLIT);
    combine_sig_kernel<<<combBlocks, 256, 0, stream>>>(part, b_global, r_sig);
    // 5) fw = w*r and its channel sums
    rowsum_kernel<<<(ROWS * VN + 7) / 8, 256, 0, stream>>>(w_sig, r_sig, fw, s2);
    // 6) refined = sum_v feats * fw_norm
    weighted_agg_kernel<<<(totalAgg + 255) / 256, 256, 0, stream>>>(feats, fw, s2, refined);
    // 7) frame projection [160 x 50176] @ [50176 x 8]
    final_proj_kernel<<<ROWS, 256, 0, stream>>>(refined, aggb, W_final, b_final, frame);
    // 8) temporal head -> [8 x 8]
    time_head_kernel<<<1, 64, 0, stream>>>(frame, W_time, b_time, out);
}
